// _ProxyFFN_57217554317808
// MI455X (gfx1250) — matrix operand order repaired and checked by reading, not compiled, not run on hardware
//
#include <hip/hip_runtime.h>
#include <stdint.h>

// ---------------- problem dims (match reference) ----------------
static constexpr int B_  = 4;
static constexpr int T_  = 2048;
static constexpr int C_  = 1024;
static constexpr int FF_ = 4096;
static constexpr int NE_ = 4;
static constexpr int N_  = B_ * T_;   // 8192 tokens

// ---------------- WMMA / TDM types ----------------
typedef __attribute__((ext_vector_type(16))) __bf16 v16bf;
typedef __attribute__((ext_vector_type(8)))  float  v8f;
typedef __attribute__((ext_vector_type(4)))  unsigned int u32x4;
typedef __attribute__((ext_vector_type(8)))  unsigned int u32x8;

union FragB16 { v16bf v; uint32_t u[8]; };

__device__ __forceinline__ unsigned short f2bf(float f) {
  union { float f; uint32_t u; } cv; cv.f = f;
  uint32_t u = cv.u;
  u += 0x7FFFu + ((u >> 16) & 1u);   // round-to-nearest-even
  return (unsigned short)(u >> 16);
}

// Low 32 bits of a generic pointer to LDS == LDS byte address (aperture form).
__device__ __forceinline__ uint32_t lds_addr_of(const void* p) {
  return (uint32_t)(uintptr_t)p;
}

// ---------------- kernel 0: x fp32 -> bf16 ----------------
__global__ __launch_bounds__(256) void cvt_bf16_kernel(
    const float* __restrict__ in, unsigned short* __restrict__ out, int n) {
  int i = blockIdx.x * 256 + threadIdx.x;
  if (i < n) out[i] = f2bf(in[i]);
}

// ---------------- kernel 1: router (one wave32 per token) ----------------
__global__ __launch_bounds__(256) void router_kernel(
    const float* __restrict__ x, const float* __restrict__ wr,
    float* __restrict__ cw) {
  const int lane = threadIdx.x & 31;
  const int tok  = (int)((blockIdx.x * blockDim.x + threadIdx.x) >> 5);
  const float* xr = x + (size_t)tok * C_;
  float a0 = 0.f, a1 = 0.f, a2 = 0.f, a3 = 0.f;
  for (int c = lane; c < C_; c += 32) {
    float xv = xr[c];
    const float* w = wr + (size_t)c * NE_;
    a0 += xv * w[0]; a1 += xv * w[1]; a2 += xv * w[2]; a3 += xv * w[3];
  }
#pragma unroll
  for (int off = 16; off >= 1; off >>= 1) {
    a0 += __shfl_xor(a0, off, 32);
    a1 += __shfl_xor(a1, off, 32);
    a2 += __shfl_xor(a2, off, 32);
    a3 += __shfl_xor(a3, off, 32);
  }
  if (lane == 0) {
    float l[4] = {a0, a1, a2, a3};
    float m = fmaxf(fmaxf(l[0], l[1]), fmaxf(l[2], l[3]));
    float w[4], s = 0.f;
#pragma unroll
    for (int i = 0; i < 4; ++i) { w[i] = __expf(l[i] - m); s += w[i]; }
    float inv = 1.f / s;
#pragma unroll
    for (int i = 0; i < 4; ++i) w[i] *= inv;
    int i1 = 0;
#pragma unroll
    for (int i = 1; i < 4; ++i) if (w[i] > w[i1]) i1 = i;
    int i2 = (i1 == 0) ? 1 : 0;
#pragma unroll
    for (int i = 0; i < 4; ++i) if (i != i1 && w[i] > w[i2]) i2 = i;
#pragma unroll
    for (int i = 0; i < 4; ++i)
      cw[(size_t)tok * NE_ + i] = (i == i1 || i == i2) ? w[i] : 0.f;
  }
}

// ---------------- kernel 2: H = silu(X@Wg) * (X@Wu), one expert ----------------
// Block tile: 128 tokens x 64 FF cols; 8 waves, each wave 16 tokens x 64 cols.
// X tile staged with the Tensor Data Mover (one descriptor per K-step).
__global__ __launch_bounds__(256) void gateup_kernel(
    const unsigned short* __restrict__ xbf,   // [N,C] bf16
    const float* __restrict__ wg,             // [C,FF] fp32 (this expert)
    const float* __restrict__ wu,             // [C,FF] fp32 (this expert)
    unsigned short* __restrict__ Hout) {      // [N,FF] bf16
  __shared__ __align__(16) unsigned short Xs[128][32];
  __shared__ __align__(16) unsigned short Gs[64][34];   // padded, transposed [f][k]
  __shared__ __align__(16) unsigned short Us[64][34];

  const int f0   = blockIdx.x * 64;
  const int n0   = blockIdx.y * 128;
  const int tid  = threadIdx.x;
  const int wv   = tid >> 5;
  const int lane = tid & 31;
  const int hi   = lane >> 4;     // half-wave index
  const int ln   = lane & 15;

  const v8f vzero = {0.f, 0.f, 0.f, 0.f, 0.f, 0.f, 0.f, 0.f};
  v8f accg[4], accu[4];
#pragma unroll
  for (int i = 0; i < 4; ++i) { accg[i] = vzero; accu[i] = vzero; }

  // TDM descriptor group1 (constant across K-steps):
  //   data_size=2B, tensor_dim0=C, tensor_dim1=N, tile=32x128, dim0_stride=C
  const u32x8 g1 = { 0x00010000u,                 // wg_mask=0, data_size=1 (2B)
                     ((unsigned)C_ & 0xFFFFu) << 16,   // tensor_dim0 lo16 -> [63:48]
                     ((unsigned)N_ & 0xFFFFu) << 16,   // dim0 hi16=0 | tensor_dim1 lo16
                     32u << 16,                   // tensor_dim1 hi16=0 | tile_dim0=32
                     128u,                        // tile_dim1=128, tile_dim2=0
                     (unsigned)C_,                // tensor_dim0_stride lo32
                     0u, 0u };                    // stride hi16, dim1_stride=0
  const uint32_t xs_lds = lds_addr_of(&Xs[0][0]);

  for (int k0 = 0; k0 < C_; k0 += 32) {
    // ---- stage X tile (bf16, 128x32 = 8KB) via Tensor Data Mover ----
    if (wv == 0) {  // uniform per-wave branch: one DMA per workgroup
      uint64_t gaddr = (uint64_t)(uintptr_t)xbf +
                       ((uint64_t)n0 * C_ + (uint64_t)k0) * 2u;
      u32x4 g0 = { 1u,                              // count=1, user mode
                   xs_lds,                          // lds_addr
                   (unsigned)(gaddr & 0xFFFFFFFFu), // global_addr[31:0]
                   (unsigned)((gaddr >> 32) & 0x1FFFFFFu) | (2u << 30) }; // type=2
      asm volatile("tensor_load_to_lds %0, %1" :: "s"(g0), "s"(g1) : "memory");
      __builtin_amdgcn_s_wait_tensorcnt(0);
    }
    // ---- stage Wg/Wu tiles 32x64, convert fp32->bf16, transpose into [f][k] ----
#pragma unroll
    for (int i = 0; i < 8; ++i) {
      int idx = tid + i * 256;
      int kk = idx >> 6, ff = idx & 63;
      size_t g = (size_t)(k0 + kk) * FF_ + (f0 + ff);
      Gs[ff][kk] = f2bf(wg[g]);
      Us[ff][kk] = f2bf(wu[g]);
    }
    if (k0 + 32 < C_) {  // uniform branch: prefetch next K-slab
      __builtin_prefetch(wg + (size_t)(k0 + 32 + (tid >> 3)) * FF_ + f0, 0, 0);
      __builtin_prefetch(wu + (size_t)(k0 + 32 + (tid >> 3)) * FF_ + f0, 0, 0);
    }
    __syncthreads();

    // A fragment: row m = wv*16+ln ; element i -> K = hi*16 + i
    FragB16 a;
    const unsigned short* xrow = &Xs[wv * 16 + ln][0];
#pragma unroll
    for (int j = 0; j < 8; ++j) {
      int kk = hi * 16 + 2 * j;
      a.u[j] = *(const uint32_t*)(xrow + kk);
    }
#pragma unroll
    for (int fb = 0; fb < 4; ++fb) {
      // B fragment: col n = fb*16+ln ; element i -> K = hi*16 + i
      FragB16 bg, bu;
      const unsigned short* grow = &Gs[fb * 16 + ln][0];
      const unsigned short* urow = &Us[fb * 16 + ln][0];
#pragma unroll
      for (int j = 0; j < 8; ++j) {
        int kk = hi * 16 + 2 * j;
        bg.u[j] = *(const uint32_t*)(grow + kk);
        bu.u[j] = *(const uint32_t*)(urow + kk);
      }
      accg[fb] = __builtin_amdgcn_wmma_f32_16x16x32_bf16(
          false, a.v, false, bg.v, (short)0, accg[fb], false, false);
      accu[fb] = __builtin_amdgcn_wmma_f32_16x16x32_bf16(
          false, a.v, false, bu.v, (short)0, accu[fb], false, false);
    }
    __syncthreads();
  }

  // epilogue: h = silu(g) * u ; D element r -> M = r + hi*8, N = ln
#pragma unroll
  for (int fb = 0; fb < 4; ++fb) {
#pragma unroll
    for (int r = 0; r < 8; ++r) {
      float g = accg[fb][r];
      float u = accu[fb][r];
      float h = (g / (1.f + __expf(-g))) * u;
      int tok = n0 + wv * 16 + r + hi * 8;
      int f   = f0 + fb * 16 + ln;
      Hout[(size_t)tok * FF_ + f] = f2bf(h);
    }
  }
}

// ---------------- kernel 3: out (+)= cw_e * (H @ Wd), one expert ----------------
// H tile staged with async global->LDS copies (ASYNCcnt path, no VGPR bounce).
__global__ __launch_bounds__(256) void down_kernel(
    const unsigned short* __restrict__ H,   // [N,FF] bf16
    const float* __restrict__ wd,           // [FF,C] fp32 (this expert)
    const float* __restrict__ cw,           // [N,NE]
    float* __restrict__ out,                // [N,C] fp32
    int e, int accumulate) {
  __shared__ __align__(16) unsigned short Hs[128][32];
  __shared__ __align__(16) unsigned short Ds[64][34];

  const int c0   = blockIdx.x * 64;
  const int n0   = blockIdx.y * 128;
  const int tid  = threadIdx.x;
  const int wv   = tid >> 5;
  const int lane = tid & 31;
  const int hi   = lane >> 4;
  const int ln   = lane & 15;

  const v8f vzero = {0.f, 0.f, 0.f, 0.f, 0.f, 0.f, 0.f, 0.f};
  v8f acc[4];
#pragma unroll
  for (int i = 0; i < 4; ++i) acc[i] = vzero;

  const uint32_t hs_lds = lds_addr_of(&Hs[0][0]);

  for (int k0 = 0; k0 < FF_; k0 += 32) {
    // ---- stage H tile (128x32 bf16) via async global->LDS b128 copies ----
    // 512 chunks of 16B; 2 per thread. chunk -> row = chunk/4, 16B-col = chunk%4
#pragma unroll
    for (int i = 0; i < 2; ++i) {
      int chunk = tid + i * 256;
      int r  = chunk >> 2;
      int c4 = chunk & 3;
      uint32_t voff = (uint32_t)((n0 + r) * FF_ + k0) * 2u + (uint32_t)c4 * 16u;
      uint32_t ldst = hs_lds + (uint32_t)r * 64u + (uint32_t)c4 * 16u;
      asm volatile("global_load_async_to_lds_b128 %0, %1, %2 offset:0"
                   :: "v"(ldst), "v"(voff), "s"(H) : "memory");
    }
    // ---- stage Wd tile 32x64, convert fp32->bf16, transpose into [c][k] ----
#pragma unroll
    for (int i = 0; i < 8; ++i) {
      int idx = tid + i * 256;
      int kk = idx >> 6, cc = idx & 63;
      Ds[cc][kk] = f2bf(wd[(size_t)(k0 + kk) * C_ + (c0 + cc)]);
    }
    if (k0 + 32 < FF_) {
      __builtin_prefetch(wd + (size_t)(k0 + 32 + (tid >> 3)) * C_ + c0, 0, 0);
    }
    asm volatile("s_wait_asynccnt 0x0" ::: "memory");
    __syncthreads();

    FragB16 a;
    const unsigned short* hrow = &Hs[wv * 16 + ln][0];
#pragma unroll
    for (int j = 0; j < 8; ++j) {
      int kk = hi * 16 + 2 * j;
      a.u[j] = *(const uint32_t*)(hrow + kk);
    }
#pragma unroll
    for (int fb = 0; fb < 4; ++fb) {
      FragB16 bd;
      const unsigned short* drow = &Ds[fb * 16 + ln][0];
#pragma unroll
      for (int j = 0; j < 8; ++j) {
        int kk = hi * 16 + 2 * j;
        bd.u[j] = *(const uint32_t*)(drow + kk);
      }
      acc[fb] = __builtin_amdgcn_wmma_f32_16x16x32_bf16(
          false, a.v, false, bd.v, (short)0, acc[fb], false, false);
    }
    __syncthreads();
  }

#pragma unroll
  for (int fb = 0; fb < 4; ++fb) {
#pragma unroll
    for (int r = 0; r < 8; ++r) {
      int tok = n0 + wv * 16 + r + hi * 8;
      int c   = c0 + fb * 16 + ln;
      float v = cw[(size_t)tok * NE_ + e] * acc[fb][r];
      size_t oi = (size_t)tok * C_ + c;
      if (accumulate) out[oi] += v;
      else            out[oi]  = v;
    }
  }
}

// ---------------- launch ----------------
extern "C" void kernel_launch(void* const* d_in, const int* in_sizes, int n_in,
                              void* d_out, int out_size, void* d_ws, size_t ws_size,
                              hipStream_t stream) {
  (void)in_sizes; (void)n_in; (void)out_size; (void)ws_size;
  const float* x  = (const float*)d_in[0];
  const float* wr = (const float*)d_in[1];
  const float* wg = (const float*)d_in[2];
  const float* wu = (const float*)d_in[3];
  const float* wd = (const float*)d_in[4];
  float* out = (float*)d_out;

  // workspace layout: xbf16 (16 MB) | cw (128 KB) | H bf16 (64 MB)
  char* ws = (char*)d_ws;
  unsigned short* xbf = (unsigned short*)ws;
  size_t off = ((size_t)N_ * C_ * sizeof(unsigned short) + 255) & ~(size_t)255;
  float* cw = (float*)(ws + off);
  off += ((size_t)N_ * NE_ * sizeof(float) + 255) & ~(size_t)255;
  unsigned short* Hbuf = (unsigned short*)(ws + off);

  const int nc = N_ * C_;
  hipLaunchKernelGGL(cvt_bf16_kernel, dim3((nc + 255) / 256), dim3(256), 0, stream,
                     x, xbf, nc);
  hipLaunchKernelGGL(router_kernel, dim3(N_ / 8), dim3(256), 0, stream, x, wr, cw);

  for (int e = 0; e < NE_; ++e) {
    hipLaunchKernelGGL(gateup_kernel, dim3(FF_ / 64, N_ / 128), dim3(256), 0, stream,
                       xbf, wg + (size_t)e * C_ * FF_, wu + (size_t)e * C_ * FF_, Hbuf);
    hipLaunchKernelGGL(down_kernel, dim3(C_ / 64, N_ / 128), dim3(256), 0, stream,
                       Hbuf, wd + (size_t)e * FF_ * C_, cw, out, e, (e > 0) ? 1 : 0);
  }
}
